// Capsule_network_47631187312647
// MI455X (gfx1250) — matrix operand order repaired and checked by reading, not compiled, not run on hardware
//
#include <hip/hip_runtime.h>
#include <hip/hip_bf16.h>

// ---------------------------------------------------------------------------
// CapsNet forward on MI455X (gfx1250, wave32, WMMA + async global->LDS).
//
//   1. conv1 (direct f32 VALU, 4.2 GFLOP)  -> h1 f16 [256,256,20,20] (52MB, L2-resident)
//   2. conv2_w OIHW f32 -> f16 (layout is already [n][k] for the implicit GEMM)
//   3. conv2 implicit GEMM, v_wmma_f32_16x16x32_f16, M=9216 N=256 K=20736 (98 GFLOP),
//      double-buffered: async B tiles (GLOBAL_LOAD_ASYNC_TO_LDS_B128, ASYNCcnt) +
//      register-staged im2col A gather with incremental k-decode.
//   4. squash over width axis -> u [256,1152,8] f32
//   5. u_hat[b][i][o][d]  (189MB f32, coalesced layout for routing)
//   6. dynamic routing (3 iters) in LDS per batch element, deterministic reduction
//   7. decoder: 3x f16-WMMA GEMM (async A/B tile loads), fused bias+ReLU / sigmoid
// ---------------------------------------------------------------------------

typedef __attribute__((ext_vector_type(16))) _Float16 v16h;
typedef __attribute__((ext_vector_type(8)))  _Float16 v8h;
typedef __attribute__((ext_vector_type(8)))  float    v8f;
typedef __attribute__((ext_vector_type(4)))  int      v4i;

#define LDA 40   // padded LDS row stride (halves): 80B rows, 16B-aligned fragments
#define LDB 40

#define CAPS_EPS 1e-7f

#if defined(__gfx1250__) && \
    __has_builtin(__builtin_amdgcn_global_load_async_to_lds_b128) && \
    __has_builtin(__builtin_amdgcn_s_wait_asynccnt)
#define CDNA5_HAS_ASYNC 1
#else
#define CDNA5_HAS_ASYNC 0
#endif

#define AS_LDS __attribute__((address_space(3)))

// 16B global -> LDS copy; async (ASYNCcnt-tracked, no VGPR round trip) when available.
// Builtin signature (probed via compiler diagnostic): param0 = v4i* (global side,
// generic AS), param1 = LDS-side v4i*, then imm offset, imm cpol.
__device__ __forceinline__ void cp16_g2l(const _Float16* g, _Float16* l) {
#if CDNA5_HAS_ASYNC
  __builtin_amdgcn_global_load_async_to_lds_b128((v4i*)g, (AS_LDS v4i*)l, 0, 0);
#else
  *(v8h*)l = *(const v8h*)g;
#endif
}

__device__ __forceinline__ void wait_async_all() {
#if CDNA5_HAS_ASYNC
  __builtin_amdgcn_s_wait_asynccnt(0);
#endif
}

// ---- WMMA fragment helpers (A and B: contiguous k, lane half h holds k 16h..16h+15) ------

__device__ __forceinline__ v16h frag_from(const _Float16* p0, const _Float16* p1) {
  v8h lo = *(const v8h*)p0;
  v8h hi = *(const v8h*)p1;
  return __builtin_shufflevector(lo, hi, 0,1,2,3,4,5,6,7,8,9,10,11,12,13,14,15);
}

// A tile in LDS: As[row][k].  Lane<16: K0..15, lane>=16: K16..31.
__device__ __forceinline__ v16h load_a_frag(const _Float16* As, int rowBase, int lane) {
  int r  = rowBase + (lane & 15);
  int ak = (lane & 16) ? 16 : 0;
  const _Float16* base = As + r * LDA;
  return frag_from(base + ak, base + ak + 8);
}

// B tile in LDS stored [n][k].  Lane<16: N=lane K0..15, lane>=16: N=lane-16 K16..31.
__device__ __forceinline__ v16h load_b_frag(const _Float16* Bs, int colBase, int lane) {
  int c  = colBase + (lane & 15);
  int kb = (lane & 16) ? 16 : 0;
  const _Float16* base = Bs + c * LDB + kb;
  return frag_from(base, base + 8);
}

__device__ __forceinline__ v8f wmma_f16(v16h a, v16h b, v8f c) {
  return __builtin_amdgcn_wmma_f32_16x16x32_f16(false, a, false, b, (short)0, c,
                                                false, false);
}

// ---------------------------------------------------------------------------
// 1) conv1: [256,1,28,28] * [256,1,9,9] -> h1 f16 [256,256,20,20]
// ---------------------------------------------------------------------------
__global__ void conv1_kernel(const float* __restrict__ x,
                             const float* __restrict__ w,
                             const float* __restrict__ bias,
                             _Float16* __restrict__ h1) {
  __shared__ float xs[784];
  __shared__ float ws[16 * 81];
  const int b = blockIdx.x, ocg = blockIdx.y, t = threadIdx.x;
  for (int i = t; i < 784; i += 256)  xs[i] = x[(size_t)b * 784 + i];
  for (int i = t; i < 1296; i += 256) ws[i] = w[(size_t)ocg * 16 * 81 + i];
  __syncthreads();
  const int ocl = t >> 4, p = t & 15;
  const int oc = ocg * 16 + ocl;
  const float bv = bias[oc];
  const float* wr = ws + ocl * 81;
  for (int j = 0; j < 25; ++j) {
    int pos = p * 25 + j;
    int oy = pos / 20, ox = pos - oy * 20;
    const float* xr = xs + oy * 28 + ox;
    float acc = bv;
#pragma unroll
    for (int ky = 0; ky < 9; ++ky)
#pragma unroll
      for (int kx = 0; kx < 9; ++kx)
        acc += xr[ky * 28 + kx] * wr[ky * 9 + kx];
    h1[((((size_t)b * 256 + oc) * 20 + oy) * 20) + ox] = (_Float16)acc;
  }
}

__global__ void f32_to_f16_kernel(const float* __restrict__ src,
                                  _Float16* __restrict__ dst, int n) {
  int idx = blockIdx.x * 256 + threadIdx.x;
  if (idx < n) dst[idx] = (_Float16)src[idx];
}

// dst[n*K + k] = (f16) src[k*N + n]   (decoder weights: [K][N] -> [N][K])
__global__ void transpose_f32_to_f16_kernel(const float* __restrict__ src,
                                            _Float16* __restrict__ dst,
                                            int K, int N) {
  int idx = blockIdx.x * 256 + threadIdx.x;
  if (idx >= K * N) return;
  int n = idx / K, k = idx - n * K;
  dst[idx] = (_Float16)src[(size_t)k * N + n];
}

// ---------------------------------------------------------------------------
// 3) conv2 implicit GEMM with WMMA, double buffered.
//    C[m][n] = sum_k A[m][k] * B[n][k] + bias[n]
//    m = b*36 + oy*6 + ox,  k = ic*81 + ky*9 + kx,  n = oc
//    A[m][k] = h1[b][ic][2*oy+ky][2*ox+kx]   (VALID, stride 2, always in-bounds)
//    BwT = conv2_w cast to f16; OIHW is already [n][k].
// ---------------------------------------------------------------------------
__global__ void conv2_gemm_kernel(const _Float16* __restrict__ h1,
                                  const _Float16* __restrict__ BwT,
                                  const float* __restrict__ bias,
                                  float* __restrict__ out) {
  __shared__ __align__(16) _Float16 As[2][64 * LDA];
  __shared__ __align__(16) _Float16 Bs[2][128 * LDB];
  const int t = threadIdx.x, lane = t & 31, w = t >> 5;
  const int wm = w & 1, wn = w >> 1;
  const int m0 = blockIdx.x * 64;
  const int nb = blockIdx.y * 128;
  const int K = 20736, NSTEP = K / 32;

  // --- per-thread im2col state (A): one row, 8 consecutive k slots -----------
  const int row = t >> 2;            // 0..63
  const int kk0 = (t & 3) * 8;       // 0,8,16,24
  const int m = m0 + row;
  const int bb = m / 36;
  const int rm = m - bb * 36;
  const int oy = rm / 6, ox = rm - oy * 6;
  const _Float16* abase = h1 + (size_t)bb * 102400 + oy * 40 + ox * 2;
  int ick[8], rk[8];
#pragma unroll
  for (int j = 0; j < 8; ++j) { ick[j] = 0; rk[j] = kk0 + j; }  // k0 = 0 decode

  // --- B copy geometry: 128 rows x 32 k halves; thread moves 32B ------------
  const int brow = t >> 1, bsel = t & 1;
  const _Float16* bsrc = BwT + (size_t)(nb + brow) * K + bsel * 16;

  v8f acc[2][2];
#pragma unroll
  for (int mt = 0; mt < 2; ++mt)
#pragma unroll
    for (int nt = 0; nt < 2; ++nt)
#pragma unroll
      for (int r = 0; r < 8; ++r) acc[mt][nt][r] = 0.f;

  // ---- prologue: tile 0 -----------------------------------------------------
  {
    _Float16* bl = &Bs[0][brow * LDB + bsel * 16];
    cp16_g2l(bsrc, bl);
    cp16_g2l(bsrc + 8, bl + 8);
    v8h areg;
#pragma unroll
    for (int j = 0; j < 8; ++j) {
      int ky = rk[j] / 9, kx = rk[j] - ky * 9;
      areg[j] = abase[ick[j] * 400 + ky * 20 + kx];
      rk[j] += 32;
      if (rk[j] >= 81) { rk[j] -= 81; ick[j] += 1; }
    }
    *(v8h*)(&As[0][row * LDA + kk0]) = areg;
    wait_async_all();
  }
  __syncthreads();

  // ---- main loop ------------------------------------------------------------
  for (int ks = 0; ks < NSTEP; ++ks) {
    const int cur = ks & 1, nxt = cur ^ 1;
    const bool pf = (ks + 1 < NSTEP);
    v8h areg;
    if (pf) {
      // async B prefetch for next tile (overlaps with WMMA below)
      const _Float16* gp = bsrc + (size_t)(ks + 1) * 32;
      _Float16* bl = &Bs[nxt][brow * LDB + bsel * 16];
      cp16_g2l(gp, bl);
      cp16_g2l(gp + 8, bl + 8);
      // A gather for next tile into registers (loadcnt-covered until store)
#pragma unroll
      for (int j = 0; j < 8; ++j) {
        int ky = rk[j] / 9, kx = rk[j] - ky * 9;
        areg[j] = abase[ick[j] * 400 + ky * 20 + kx];
        rk[j] += 32;
        if (rk[j] >= 81) { rk[j] -= 81; ick[j] += 1; }
      }
    }

    v16h af0 = load_a_frag(As[cur], wm * 32,      lane);
    v16h af1 = load_a_frag(As[cur], wm * 32 + 16, lane);
    v16h bf0 = load_b_frag(Bs[cur], wn * 32,      lane);
    v16h bf1 = load_b_frag(Bs[cur], wn * 32 + 16, lane);
    acc[0][0] = wmma_f16(af0, bf0, acc[0][0]);
    acc[0][1] = wmma_f16(af0, bf1, acc[0][1]);
    acc[1][0] = wmma_f16(af1, bf0, acc[1][0]);
    acc[1][1] = wmma_f16(af1, bf1, acc[1][1]);

    if (pf) {
      *(v8h*)(&As[nxt][row * LDA + kk0]) = areg;
      wait_async_all();
    }
    __syncthreads();
  }

  // ---- epilogue: D vgpr r -> M = r (+8 for lanes 16..31), N = lane&15 -------
  const int nlo = lane & 15;
  const int mhi = (lane & 16) ? 8 : 0;
#pragma unroll
  for (int mt = 0; mt < 2; ++mt)
#pragma unroll
    for (int nt = 0; nt < 2; ++nt) {
      int gm0 = m0 + wm * 32 + mt * 16 + mhi;
      int gn  = nb + wn * 32 + nt * 16 + nlo;
      float bv = bias[gn];
#pragma unroll
      for (int r = 0; r < 8; ++r)
        out[(size_t)(gm0 + r) * 256 + gn] = acc[mt][nt][r] + bv;
    }
}

// ---------------------------------------------------------------------------
// 4) squash over width axis (axis=-1, 6 elems); u flat = b*9216 + c*36 + y*6 + x
// ---------------------------------------------------------------------------
__global__ void squash_primary_kernel(const float* __restrict__ h2,
                                      float* __restrict__ u) {
  int idx = blockIdx.x * 256 + threadIdx.x;   // (b, c, y)
  if (idx >= 256 * 256 * 6) return;
  int b = idx / 1536, rem = idx - b * 1536;
  int c = rem / 6, y = rem - c * 6;
  float vbuf[6], n2 = 0.f;
#pragma unroll
  for (int x = 0; x < 6; ++x) {
    int m = b * 36 + y * 6 + x;
    float v = h2[(size_t)m * 256 + c];
    vbuf[x] = v;
    n2 += v * v;
  }
  float nr = sqrtf(n2);
  float sc = (n2 / (1.f + n2)) / (nr + CAPS_EPS);
#pragma unroll
  for (int x = 0; x < 6; ++x)
    u[(size_t)b * 9216 + c * 36 + y * 6 + x] = vbuf[x] * sc;
}

// ---------------------------------------------------------------------------
// 5) u_hat[b][i][o][d] = sum_e W_caps[o][i][d][e] * u[b][i][e]
// ---------------------------------------------------------------------------
__global__ void uhat_kernel(const float* __restrict__ W,
                            const float* __restrict__ u,
                            float* __restrict__ uhat) {
  int idx = blockIdx.x * 256 + threadIdx.x;
  if (idx >= 256 * 1152 * 10 * 16) return;
  int d = idx & 15;
  int r = idx >> 4;
  int o = r % 10;  r /= 10;
  int i = r % 1152;
  int b = r / 1152;
  const float* wp = W + (((size_t)o * 1152 + i) * 16 + d) * 8;
  const float* up = u + (size_t)b * 9216 + i * 8;
  float acc = 0.f;
#pragma unroll
  for (int e = 0; e < 8; ++e) acc += wp[e] * up[e];
  uhat[idx] = acc;
}

// ---------------------------------------------------------------------------
// 6) dynamic routing, one block per batch element, all state in LDS.
// ---------------------------------------------------------------------------
__global__ void routing_kernel(const float* __restrict__ u_hat,
                               float* __restrict__ out_len,
                               float* __restrict__ masked) {
  __shared__ float blog[10 * 1152];
  __shared__ float spart[16 * 160];
  __shared__ float s[160];
  __shared__ float vv[160];
  __shared__ float lens[10];
  __shared__ int best;
  const int b = blockIdx.x, t = threadIdx.x;
  const int dl = t & 15, grp = t >> 4;
  const float* uh = u_hat + (size_t)b * 1152 * 160;

  for (int i = t; i < 11520; i += 256) blog[i] = 0.f;
  __syncthreads();

  for (int it = 0; it < 3; ++it) {
    float sl[10];
#pragma unroll
    for (int o = 0; o < 10; ++o) sl[o] = 0.f;
    for (int i = grp; i < 1152; i += 16) {
      float lo[10], mx = -3.4e38f;
#pragma unroll
      for (int o = 0; o < 10; ++o) {
        lo[o] = blog[o * 1152 + i];
        mx = fmaxf(mx, lo[o]);
      }
      float sum = 0.f;
#pragma unroll
      for (int o = 0; o < 10; ++o) { lo[o] = __expf(lo[o] - mx); sum += lo[o]; }
      float inv = 1.f / sum;
      const float* up = uh + (size_t)i * 160 + dl;
#pragma unroll
      for (int o = 0; o < 10; ++o) sl[o] += lo[o] * inv * up[o * 16];
    }
#pragma unroll
    for (int o = 0; o < 10; ++o) spart[grp * 160 + o * 16 + dl] = sl[o];
    __syncthreads();
    if (t < 160) {
      float a = 0.f;
#pragma unroll
      for (int g = 0; g < 16; ++g) a += spart[g * 160 + t];
      s[t] = a;
    }
    __syncthreads();
    if (t < 10) {
      float n2 = 0.f;
#pragma unroll
      for (int d = 0; d < 16; ++d) n2 += s[t * 16 + d] * s[t * 16 + d];
      float nr = sqrtf(n2);
      float sc = (n2 / (1.f + n2)) / (nr + CAPS_EPS);
#pragma unroll
      for (int d = 0; d < 16; ++d) vv[t * 16 + d] = s[t * 16 + d] * sc;
    }
    __syncthreads();
    if (it < 2) {
      for (int i = t; i < 1152; i += 256) {
        const float* up = uh + (size_t)i * 160;
#pragma unroll
        for (int o = 0; o < 10; ++o) {
          float dp = 0.f;
#pragma unroll
          for (int d = 0; d < 16; ++d) dp += vv[o * 16 + d] * up[o * 16 + d];
          blog[o * 1152 + i] += dp;
        }
      }
      __syncthreads();
    }
  }

  if (t < 10) {
    float n2 = 0.f;
#pragma unroll
    for (int d = 0; d < 16; ++d) n2 += vv[t * 16 + d] * vv[t * 16 + d];
    float l = sqrtf(n2);
    lens[t] = l;
    out_len[(size_t)b * 10 + t] = l;
  }
  __syncthreads();
  if (t == 0) {
    int bi = 0; float bl = lens[0];
#pragma unroll
    for (int o = 1; o < 10; ++o)
      if (lens[o] > bl) { bl = lens[o]; bi = o; }
    best = bi;
  }
  __syncthreads();
  if (t < 160) {
    int o = t >> 4;
    masked[(size_t)b * 160 + t] = (o == best) ? vv[t] : 0.f;
  }
}

// ---------------------------------------------------------------------------
// 7) generic WMMA GEMM: A f16 [M][K] row-major, BT f16 [N][K] row-major.
//    Async tile loads; fused bias + activation.  ACT: 0 none, 1 ReLU, 2 sigmoid.
//    M % 64 == 0, K % 32 == 0, N % 16 == 0.
// ---------------------------------------------------------------------------
template <typename OutT, int ACT>
__global__ void gemm_f16_kernel(const _Float16* __restrict__ A,
                                const _Float16* __restrict__ BT,
                                const float* __restrict__ bias,
                                OutT* __restrict__ Out,
                                int M, int N, int K, int ldo) {
  __shared__ __align__(16) _Float16 As[64 * LDA];
  __shared__ __align__(16) _Float16 Bs[128 * LDB];
  const int t = threadIdx.x, lane = t & 31, w = t >> 5;
  const int wm = w & 1, wn = w >> 1;
  const int m0 = blockIdx.x * 64;
  const int nb = blockIdx.y * 128;
  const int arow = t >> 2, akc = (t & 3) * 8;   // A: 64 x 32 halves, 16B/thread
  const int brow = t >> 1, bsel = t & 1;        // B: 128 x 32 halves, 32B/thread
  const bool bvalid = (nb + brow) < N;

  v8f acc[2][2];
#pragma unroll
  for (int mt = 0; mt < 2; ++mt)
#pragma unroll
    for (int nt = 0; nt < 2; ++nt)
#pragma unroll
      for (int r = 0; r < 8; ++r) acc[mt][nt][r] = 0.f;

  for (int k0 = 0; k0 < K; k0 += 32) {
    cp16_g2l(A + (size_t)(m0 + arow) * K + k0 + akc, &As[arow * LDA + akc]);
    _Float16* bl = &Bs[brow * LDB + bsel * 16];
    if (bvalid) {
      const _Float16* gp = BT + (size_t)(nb + brow) * K + k0 + bsel * 16;
      cp16_g2l(gp, bl);
      cp16_g2l(gp + 8, bl + 8);
    } else {
      v8h z;
#pragma unroll
      for (int jj = 0; jj < 8; ++jj) z[jj] = (_Float16)0.f;
      *(v8h*)bl = z;
      *(v8h*)(bl + 8) = z;
    }
    wait_async_all();
    __syncthreads();

    v16h af0 = load_a_frag(As, wm * 32,      lane);
    v16h af1 = load_a_frag(As, wm * 32 + 16, lane);
    v16h bf0 = load_b_frag(Bs, wn * 32,      lane);
    v16h bf1 = load_b_frag(Bs, wn * 32 + 16, lane);
    acc[0][0] = wmma_f16(af0, bf0, acc[0][0]);
    acc[0][1] = wmma_f16(af0, bf1, acc[0][1]);
    acc[1][0] = wmma_f16(af1, bf0, acc[1][0]);
    acc[1][1] = wmma_f16(af1, bf1, acc[1][1]);
    __syncthreads();
  }

  const int nlo = lane & 15;
  const int mhi = (lane & 16) ? 8 : 0;
#pragma unroll
  for (int mt = 0; mt < 2; ++mt)
#pragma unroll
    for (int nt = 0; nt < 2; ++nt) {
      int gm0 = m0 + wm * 32 + mt * 16 + mhi;
      int gn  = nb + wn * 32 + nt * 16 + nlo;
      if (gn < N) {
        float bv = bias[gn];
#pragma unroll
        for (int r = 0; r < 8; ++r) {
          float v = acc[mt][nt][r] + bv;
          if (ACT == 1) v = fmaxf(v, 0.f);
          if (ACT == 2) v = 1.f / (1.f + __expf(-v));
          Out[(size_t)(gm0 + r) * ldo + gn] = (OutT)v;
        }
      }
    }
}

// ---------------------------------------------------------------------------
// Host-side launch sequence
// ---------------------------------------------------------------------------
extern "C" void kernel_launch(void* const* d_in, const int* in_sizes, int n_in,
                              void* d_out, int out_size, void* d_ws, size_t ws_size,
                              hipStream_t stream) {
  (void)in_sizes; (void)n_in; (void)out_size; (void)ws_size;
  const float* x       = (const float*)d_in[0];
  const float* conv1_w = (const float*)d_in[1];
  const float* conv1_b = (const float*)d_in[2];
  const float* conv2_w = (const float*)d_in[3];
  const float* conv2_b = (const float*)d_in[4];
  const float* W_caps  = (const float*)d_in[5];
  const float* dec_w1  = (const float*)d_in[6];
  const float* dec_b1  = (const float*)d_in[7];
  const float* dec_w2  = (const float*)d_in[8];
  const float* dec_b2  = (const float*)d_in[9];
  const float* dec_w3  = (const float*)d_in[10];
  const float* dec_b3  = (const float*)d_in[11];
  float* out = (float*)d_out;              // [256*10] lengths ++ [256*784] recon

  char* ws = (char*)d_ws;
  size_t off = 0;
  auto alloc = [&](size_t bytes) {
    char* p = ws + off;
    off += (bytes + 255) & ~(size_t)255;
    return p;
  };
  _Float16* h1f16   = (_Float16*)alloc((size_t)256 * 256 * 20 * 20 * 2); // 52.4 MB
  _Float16* BwT     = (_Float16*)alloc((size_t)256 * 20736 * 2);         // 10.6 MB
  float*    h2      = (float*)   alloc((size_t)9216 * 256 * 4);          //  9.4 MB
  float*    u       = (float*)   alloc((size_t)256 * 9216 * 4);          //  9.4 MB
  float*    u_hat   = (float*)   alloc((size_t)256 * 1152 * 160 * 4);    // 188.7 MB
  float*    masked  = (float*)   alloc((size_t)256 * 160 * 4);
  _Float16* maskedh = (_Float16*)alloc((size_t)256 * 160 * 2);
  _Float16* w1t     = (_Float16*)alloc((size_t)512 * 160 * 2);           // [N][K]
  _Float16* w2t     = (_Float16*)alloc((size_t)1024 * 512 * 2);
  _Float16* w3t     = (_Float16*)alloc((size_t)784 * 1024 * 2);
  _Float16* h1d     = (_Float16*)alloc((size_t)256 * 512 * 2);
  _Float16* h2d     = (_Float16*)alloc((size_t)256 * 1024 * 2);

  // 1) conv1 -> h1 f16
  conv1_kernel<<<dim3(256, 16), 256, 0, stream>>>(x, conv1_w, conv1_b, h1f16);

  // 2) weight conversions: conv2_w is already [n][k]; decoder weights transposed
  f32_to_f16_kernel<<<(256 * 20736) / 256, 256, 0, stream>>>(conv2_w, BwT, 256 * 20736);
  transpose_f32_to_f16_kernel<<<(160 * 512 + 255) / 256, 256, 0, stream>>>(dec_w1, w1t, 160, 512);
  transpose_f32_to_f16_kernel<<<(512 * 1024 + 255) / 256, 256, 0, stream>>>(dec_w2, w2t, 512, 1024);
  transpose_f32_to_f16_kernel<<<(1024 * 784 + 255) / 256, 256, 0, stream>>>(dec_w3, w3t, 1024, 784);

  // 3) conv2 implicit GEMM (M=9216, N=256, K=20736)
  conv2_gemm_kernel<<<dim3(9216 / 64, 256 / 128), 256, 0, stream>>>(h1f16, BwT, conv2_b, h2);

  // 4) squash over width -> u
  squash_primary_kernel<<<(256 * 256 * 6) / 256, 256, 0, stream>>>(h2, u);

  // 5) u_hat
  uhat_kernel<<<(256 * 1152 * 160) / 256, 256, 0, stream>>>(W_caps, u, u_hat);

  // 6) routing -> lengths (d_out head) + masked
  routing_kernel<<<256, 256, 0, stream>>>(u_hat, out, masked);

  // 7) decoder
  f32_to_f16_kernel<<<(256 * 160 + 255) / 256, 256, 0, stream>>>(masked, maskedh, 256 * 160);
  gemm_f16_kernel<_Float16, 1><<<dim3(256 / 64, 512 / 128), 256, 0, stream>>>(
      maskedh, w1t, dec_b1, h1d, 256, 512, 160, 512);
  gemm_f16_kernel<_Float16, 1><<<dim3(256 / 64, 1024 / 128), 256, 0, stream>>>(
      h1d, w2t, dec_b2, h2d, 256, 1024, 512, 1024);
  gemm_f16_kernel<float, 2><<<dim3(256 / 64, (784 + 127) / 128), 256, 0, stream>>>(
      h2d, w3t, dec_b3, out + 256 * 10, 256, 784, 1024, 784);
}
